// FidelityAwareMultimodalDGCNN_7121055777269
// MI455X (gfx1250) — matrix operand order repaired and checked by reading, not compiled, not run on hardware
//
#include <hip/hip_runtime.h>
#include <hip/hip_bf16.h>

typedef __attribute__((ext_vector_type(16))) _Float16 v16h;
typedef __attribute__((ext_vector_type(8)))  float    v8f;

constexpr int B   = 8;
constexpr int N   = 2048;
constexpr int H   = 256;
constexpr int KNN = 10;

#define DEV __device__ __forceinline__

DEV v8f wmma_f16(v16h a, v16h b, v8f c) {
  // D = A(16x32 f16) * B(32x16 f16) + C(16x16 f32)
  return __builtin_amdgcn_wmma_f32_16x16x32_f16(
      /*neg_a=*/false, a, /*neg_b=*/false, b,
      /*c_mod=*/(short)0, c, /*reuse_a=*/false, /*reuse_b=*/false);
}

// A fragment: 16x32 (MxK), row-major source with leading dim ld.
template <typename T>
DEV v16h load_a(const T* p, int ld) {
  const int lane = threadIdx.x & 31;
  const int m = lane & 15, kh = lane >> 4;
  v16h r;
#pragma unroll
  for (int e = 0; e < 16; ++e) {
    int k = (e < 8) ? (kh * 8 + e) : (16 + kh * 8 + (e - 8));
    r[e] = (_Float16)p[m * ld + k];
  }
  return r;
}

// B fragment: 32x16 (KxN) where the source is stored row-major as [16 x K]
// (rows = output columns n), leading dim ld.
template <typename T>
DEV v16h load_bT(const T* p, int ld) {
  const int lane = threadIdx.x & 31;
  const int n = lane & 15, kh = lane >> 4;
  v16h r;
#pragma unroll
  for (int e = 0; e < 16; ++e) {
    int k = (e < 8) ? (kh * 8 + e) : (16 + kh * 8 + (e - 8));
    r[e] = (_Float16)p[(size_t)n * ld + k];
  }
  return r;
}

// ---------------------------------------------------------------------------
// t[b,n,h] = relu( sum_d x[b,d,n] * pw[h,d] + pb[h] ); also f16 copy and ||t||^2
__global__ void proj_kernel(const float* __restrict__ x,
                            const float* __restrict__ pw,
                            const float* __restrict__ pb, int d,
                            float* __restrict__ tf32,
                            _Float16* __restrict__ tf16,
                            float* __restrict__ xx) {
  __shared__ float xcol[304];
  __shared__ float red[256];
  const int bid = blockIdx.x;
  const int b = bid >> 11, n = bid & (N - 1);
  const int tid = threadIdx.x;
  for (int dd = tid; dd < d; dd += 256)
    xcol[dd] = x[((size_t)b * d + dd) * N + n];
  __syncthreads();
  float acc = pb[tid];
  const float* wrow = pw + (size_t)tid * d;
  for (int dd = 0; dd < d; ++dd) acc = fmaf(wrow[dd], xcol[dd], acc);
  acc = fmaxf(acc, 0.0f);
  const size_t o = ((size_t)(b * N + n)) * H + tid;
  tf32[o] = acc;
  tf16[o] = (_Float16)acc;
  red[tid] = acc * acc;
  __syncthreads();
  for (int s = 128; s > 0; s >>= 1) {
    if (tid < s) red[tid] += red[tid + s];
    __syncthreads();
  }
  if (tid == 0) xx[b * N + n] = red[0];
}

// ---------------------------------------------------------------------------
// Fused Gram + top-k: block owns (b, itile). A fragments stay in VGPRs for the
// whole j sweep. Ranking uses 2*x_i.x_j - ||x_j||^2 (the -||x_i||^2 term is a
// per-row constant and cannot change the argsort). pd is never materialized.
__global__ void gramtopk_kernel(const _Float16* __restrict__ tf16,
                                const float* __restrict__ xx,
                                int* __restrict__ knn) {
  __shared__ float sT[16 * 128];           // 8 KB score slab: 16 rows x 8 jtiles
  __shared__ float cv[32 * KNN];
  __shared__ int   ci[32 * KNN];

  const int bid = blockIdx.x;              // B * 128 itiles
  const int b = bid >> 7;
  const int itile = bid & 127;
  const int i0 = itile * 16;
  const int tid = threadIdx.x;
  const int wave = tid >> 5, lane = tid & 31;
  const int kh = lane >> 4, ln = lane & 15;

  // keep the full A row-tile (16x256 f16) resident in VGPRs
  v16h afrag[8];
#pragma unroll
  for (int kc = 0; kc < H / 32; ++kc)
    afrag[kc] = load_a(tf16 + ((size_t)(b * N + i0)) * H + kc * 32, H);

  // per-half-row running top-10 (threads 0..31: row = tid&15, half = tid>>4)
  float tv[KNN];
  int   ti[KNN];
#pragma unroll
  for (int q = 0; q < KNN; ++q) { tv[q] = -3.0e38f; ti[q] = 0; }
  const int trow = tid & 15, thalf = tid >> 4;

  for (int it = 0; it < 16; ++it) {
    const int jtile = it * 8 + wave;
    const int j0 = jtile * 16;
    v8f acc = {};
#pragma unroll
    for (int kc = 0; kc < H / 32; ++kc) {
      v16h bb = load_bT(tf16 + ((size_t)(b * N + j0)) * H + kc * 32, H);
      acc = wmma_f16(afrag[kc], bb, acc);
    }
    const float xxj = xx[b * N + j0 + ln];
#pragma unroll
    for (int rr = 0; rr < 8; ++rr) {
      const int row = rr + kh * 8;
      sT[row * 128 + wave * 16 + ln] = 2.0f * acc[rr] - xxj;
    }
    __syncthreads();
    if (tid < 32) {
      const int jbase = it * 128 + thalf * 64;
      for (int c = 0; c < 64; ++c) {
        const float v = sT[trow * 128 + thalf * 64 + c];
        if (v > tv[KNN - 1]) {
          int q = KNN - 1;
          while (q > 0 && tv[q - 1] < v) { tv[q] = tv[q - 1]; ti[q] = ti[q - 1]; --q; }
          tv[q] = v; ti[q] = jbase + c;
        }
      }
    }
    __syncthreads();
  }

  if (tid < 32) {
#pragma unroll
    for (int q = 0; q < KNN; ++q) { cv[tid * KNN + q] = tv[q]; ci[tid * KNN + q] = ti[q]; }
  }
  __syncthreads();
  if (tid < 16) {
    // merge the two half-row lists (20 candidates) -> top 10 for this row
    const int la = tid * KNN, lb = (tid + 16) * KNN;
    for (int s = 0; s < KNN; ++s) {
      int best = la; float bv = cv[la];
      for (int q = 1; q < KNN; ++q)
        if (cv[la + q] > bv) { bv = cv[la + q]; best = la + q; }
      for (int q = 0; q < KNN; ++q)
        if (cv[lb + q] > bv) { bv = cv[lb + q]; best = lb + q; }
      knn[((size_t)(b * N + i0 + tid)) * KNN + s] = ci[best];
      cv[best] = -3.0e38f;
    }
  }
}

// ---------------------------------------------------------------------------
// EdgeConv: async-stage 2 center rows to LDS, gather 2x10 neighbors ->
// edge[20x512] (padded to 32 rows), h1 = relu(edge*w1^T*bn + b1) [WMMA],
// h2 = h1*w2^T [WMMA], max over k, atomic mean over n.
__global__ void edge_kernel(const float* __restrict__ tf32,
                            const int* __restrict__ knn,
                            const float* __restrict__ w1,
                            const float* __restrict__ g1,
                            const float* __restrict__ b1,
                            const float* __restrict__ w2,
                            float* __restrict__ meanacc) {
  __shared__ _Float16 sA[32 * 512];    // 32 KB: edge tile (f16), reused as h2 (f32)
  __shared__ _Float16 sH1[32 * 256];   // 16 KB
  __shared__ float    sXC[2 * 256];    // 2 KB: center-point rows (async staged)
  float* sH2 = reinterpret_cast<float*>(sA);

  const int bid = blockIdx.x;
  const int b  = bid >> 10;
  const int n0 = (bid & 1023) * 2;
  const int tid = threadIdx.x;

  // warm L2 for the (heavily reused) weight matrices
  if (tid == 0) {
    __builtin_prefetch(w1, 0, 1);
    __builtin_prefetch(w2, 0, 1);
  }

  // async copy the two center rows (contiguous 512 floats) into LDS
  {
    const float* src = tf32 + ((size_t)(b * N + n0)) * H;
    for (int i = tid; i < 2 * H; i += 256) {
      unsigned ldsoff = (unsigned)(uintptr_t)(&sXC[i]);
      const float* g = src + i;
      asm volatile("global_load_async_to_lds_b32 %0, %1, off"
                   :: "v"(ldsoff), "v"(g) : "memory");
    }
    asm volatile("s_wait_asynccnt 0x0" ::: "memory");
  }
  __syncthreads();

  // gather edge features
  for (int e = tid; e < 32 * 512; e += 256) {
    const int row = e >> 9, col = e & 511;
    _Float16 val = (_Float16)0.0f;
    if (row < 2 * KNN) {
      const int p = row / KNN, k = row - p * KNN;
      const int n = n0 + p;
      const int c = col & 255;
      const float xc = sXC[p * 256 + c];
      if (col < 256) {
        val = (_Float16)xc;
      } else {
        const int nbr = knn[((size_t)(b * N + n)) * KNN + k];
        val = (_Float16)(tf32[((size_t)(b * N + nbr)) * H + c] - xc);
      }
    }
    sA[e] = val;
  }
  __syncthreads();

  const int wave = tid >> 5, lane = tid & 31;
  const int kh = lane >> 4, ln = lane & 15;
  const float bnscale = rsqrtf(1.0f + 1e-5f);

  // GEMM1: [32x512] x [512x256] with BN+ReLU epilogue -> sH1 (f16)
  for (int t = wave; t < 32; t += 8) {
    const int rt = t >> 4, ct = t & 15;
    v8f acc = {};
    for (int kc = 0; kc < 512 / 32; ++kc) {
      v16h a  = load_a (sA + rt * 16 * 512 + kc * 32, 512);
      v16h bb = load_bT(w1 + ((size_t)ct * 16) * 512 + kc * 32, 512);
      acc = wmma_f16(a, bb, acc);
    }
    const int ch = ct * 16 + ln;
    const float s = g1[ch] * bnscale, bia = b1[ch];
#pragma unroll
    for (int rr = 0; rr < 8; ++rr) {
      const int m = rt * 16 + rr + kh * 8;
      sH1[m * 256 + ch] = (_Float16)fmaxf(acc[rr] * s + bia, 0.0f);
    }
  }
  __syncthreads();

  // GEMM2: [32x256] x [256x256] -> sH2 (f32, overlays sA)
  for (int t = wave; t < 32; t += 8) {
    const int rt = t >> 4, ct = t & 15;
    v8f acc = {};
    for (int kc = 0; kc < 256 / 32; ++kc) {
      v16h a  = load_a (sH1 + rt * 16 * 256 + kc * 32, 256);
      v16h bb = load_bT(w2 + ((size_t)ct * 16) * 256 + kc * 32, 256);
      acc = wmma_f16(a, bb, acc);
    }
    const int ch = ct * 16 + ln;
#pragma unroll
    for (int rr = 0; rr < 8; ++rr) {
      const int m = rt * 16 + rr + kh * 8;
      sH2[m * 256 + ch] = acc[rr];
    }
  }
  __syncthreads();

  // max over k, mean over n (atomic accumulate)
  for (int e = tid; e < 2 * 256; e += 256) {
    const int p = e >> 8, ch = e & 255;
    float mx = -3.0e38f;
#pragma unroll
    for (int k = 0; k < KNN; ++k)
      mx = fmaxf(mx, sH2[(p * KNN + k) * 256 + ch]);
    atomicAdd(&meanacc[b * H + ch], mx * (1.0f / N));
  }
}

// ---------------------------------------------------------------------------
// fidelity scores + softmax + 3-layer fusion MLP, single block
__global__ void fuse_kernel(const float* __restrict__ means,
                            const float* __restrict__ U_t, const float* __restrict__ V_t,
                            const float* __restrict__ U_a, const float* __restrict__ V_a,
                            const float* __restrict__ U_v, const float* __restrict__ V_v,
                            const float* __restrict__ fw1, const float* __restrict__ fb1,
                            const float* __restrict__ fw2, const float* __restrict__ fb2,
                            const float* __restrict__ fw3, const float* __restrict__ fb3,
                            float* __restrict__ out) {
  __shared__ float sw[3];
  __shared__ float f[8 * 768];
  __shared__ float h1[8 * 384];
  __shared__ float h2[8 * 192];
  const int tid = threadIdx.x;

  if (tid == 0) {
    const float* Us[3] = {U_t, U_a, U_v};
    const float* Vs[3] = {V_t, V_a, V_v};
    float s[3];
    for (int m = 0; m < 3; ++m) {
      float uu = 0.f, vv = 0.f, uv = 0.f;
      for (int i = 0; i < 64; ++i) {
        uu += Us[m][i] * Us[m][i];
        vv += Vs[m][i] * Vs[m][i];
        uv += Us[m][i] * Vs[m][i];
      }
      const float nu = sqrtf(uu) * sqrtf(vv);
      const float mu = 0.5f + 0.5f * uv / nu;
      const float alpha = mu * nu, beta = (1.0f - mu) * nu;
      const float mean = alpha / (alpha + beta);
      const float var = alpha * beta /
          ((alpha + beta) * (alpha + beta) * (alpha + beta + 1.0f));
      s[m] = mean * rsqrtf(var);
    }
    const float mx = fmaxf(s[0], fmaxf(s[1], s[2]));
    const float e0 = expf(s[0] - mx), e1 = expf(s[1] - mx), e2 = expf(s[2] - mx);
    const float den = e0 + e1 + e2;
    sw[0] = e0 / den; sw[1] = e1 / den; sw[2] = e2 / den;
  }
  __syncthreads();

  for (int e = tid; e < 8 * 768; e += 256) {
    const int bi = e / 768, c = e - bi * 768;
    const int m = c >> 8;
    f[e] = sw[m] * means[m * (B * H) + bi * H + (c & 255)];
  }
  __syncthreads();
  for (int o = tid; o < 8 * 384; o += 256) {
    const int bi = o / 384, j = o - bi * 384;
    float acc = fb1[j];
    const float* wr = fw1 + (size_t)j * 768;
    for (int k = 0; k < 768; ++k) acc = fmaf(f[bi * 768 + k], wr[k], acc);
    h1[o] = fmaxf(acc, 0.0f);
  }
  __syncthreads();
  for (int o = tid; o < 8 * 192; o += 256) {
    const int bi = o / 192, j = o - bi * 192;
    float acc = fb2[j];
    const float* wr = fw2 + (size_t)j * 384;
    for (int k = 0; k < 384; ++k) acc = fmaf(h1[bi * 384 + k], wr[k], acc);
    h2[o] = fmaxf(acc, 0.0f);
  }
  __syncthreads();
  if (tid < 8) {
    float acc = fb3[0];
    for (int k = 0; k < 192; ++k) acc = fmaf(h2[tid * 192 + k], fw3[k], acc);
    out[tid] = acc;
  }
}

__global__ void zero_kernel(float* __restrict__ p, int n) {
  const int i = blockIdx.x * 256 + threadIdx.x;
  if (i < n) p[i] = 0.0f;
}

// ---------------------------------------------------------------------------
extern "C" void kernel_launch(void* const* d_in, const int* in_sizes, int n_in,
                              void* d_out, int out_size, void* d_ws, size_t ws_size,
                              hipStream_t stream) {
  (void)in_sizes; (void)n_in; (void)out_size; (void)ws_size;

  const float* xs[3]  = {(const float*)d_in[0], (const float*)d_in[1], (const float*)d_in[2]};
  const int    dims[3] = {300, 74, 35};
  const float* pw[3] = {(const float*)d_in[3], (const float*)d_in[5], (const float*)d_in[7]};
  const float* pb[3] = {(const float*)d_in[4], (const float*)d_in[6], (const float*)d_in[8]};
  const float* w1[3] = {(const float*)d_in[9],  (const float*)d_in[13], (const float*)d_in[17]};
  const float* g1[3] = {(const float*)d_in[10], (const float*)d_in[14], (const float*)d_in[18]};
  const float* b1[3] = {(const float*)d_in[11], (const float*)d_in[15], (const float*)d_in[19]};
  const float* w2[3] = {(const float*)d_in[12], (const float*)d_in[16], (const float*)d_in[20]};
  const float* U_t = (const float*)d_in[21];
  const float* V_t = (const float*)d_in[22];
  const float* U_a = (const float*)d_in[23];
  const float* V_a = (const float*)d_in[24];
  const float* U_v = (const float*)d_in[25];
  const float* V_v = (const float*)d_in[26];
  const float* fw1 = (const float*)d_in[27];
  const float* fb1 = (const float*)d_in[28];
  const float* fw2 = (const float*)d_in[29];
  const float* fb2 = (const float*)d_in[30];
  const float* fw3 = (const float*)d_in[31];
  const float* fb3 = (const float*)d_in[32];

  // workspace carve-up (buffers reused across modalities; no pd matrix anymore)
  char* wsp = (char*)d_ws;
  auto take = [&](size_t bytes) -> char* {
    char* p = wsp;
    wsp += (bytes + 255) & ~(size_t)255;
    return p;
  };
  float*     tf32  = (float*)    take(sizeof(float)    * (size_t)B * N * H);
  _Float16*  tf16  = (_Float16*) take(sizeof(_Float16) * (size_t)B * N * H);
  float*     xx    = (float*)    take(sizeof(float)    * (size_t)B * N);
  int*       knn   = (int*)      take(sizeof(int)      * (size_t)B * N * KNN);
  float*     means = (float*)    take(sizeof(float)    * 3 * B * H);

  zero_kernel<<<(3 * B * H + 255) / 256, 256, 0, stream>>>(means, 3 * B * H);

  for (int m = 0; m < 3; ++m) {
    proj_kernel<<<B * N, 256, 0, stream>>>(xs[m], pw[m], pb[m], dims[m], tf32, tf16, xx);
    gramtopk_kernel<<<B * (N / 16), 256, 0, stream>>>(tf16, xx, knn);
    edge_kernel<<<B * N / 2, 256, 0, stream>>>(tf32, knn, w1[m], g1[m], b1[m], w2[m],
                                               means + m * (B * H));
  }

  fuse_kernel<<<1, 256, 0, stream>>>(means, U_t, V_t, U_a, V_a, U_v, V_v,
                                     fw1, fb1, fw2, fb2, fw3, fb3, (float*)d_out);
}
